// HARMamba_SingleSensor_v1_31851477467335
// MI455X (gfx1250) — matrix operand order repaired and checked by reading, not compiled, not run on hardware
//
#include <hip/hip_runtime.h>
#include <hip/hip_bf16.h>

// ---------------- Model constants ----------------
#define kB     16
#define kL     2048
#define kCIN   7
#define kE     128
#define kDEPTH 12
#define kPATCH 16
#define kP     128            // (2048-16)/16+1
#define kDI    256
#define kDS    16
#define kDR    8
#define kDCONV 4
#define kNCLS  18
#define kBP    (kB * kP)      // 2048
#define kEPS   1e-5f

typedef _Float16 v16h __attribute__((ext_vector_type(16)));
typedef _Float16 h8v  __attribute__((ext_vector_type(8)));
typedef float    v8f  __attribute__((ext_vector_type(8)));

__device__ __forceinline__ float siluf(float x) { return x / (1.0f + __expf(-x)); }
__device__ __forceinline__ float softplusf(float x) { return (x > 20.0f) ? x : log1pf(__expf(x)); }

// ---------------- f32 -> f16 conversion ----------------
__global__ void cvt16_kernel(const float* __restrict__ src, _Float16* __restrict__ dst, int n) {
    int i = blockIdx.x * 256 + threadIdx.x;
    if (i < n) dst[i] = (_Float16)src[i];
}

// ---------------- RevIN statistics: per (b,c) mean / rstd over L ----------------
__global__ void revin_stats_kernel(const float* __restrict__ imu, float* __restrict__ mu_rstd) {
    int bc = blockIdx.x;                 // 0 .. B*CIN-1
    int b = bc / kCIN, c = bc % kCIN;
    float s = 0.f, ss = 0.f;
    for (int i = threadIdx.x; i < kL; i += 256) {
        float v = imu[((size_t)b * kL + i) * kCIN + c];
        s += v; ss += v * v;
    }
    __shared__ float s1[256], s2[256];
    s1[threadIdx.x] = s; s2[threadIdx.x] = ss;
    __syncthreads();
    for (int st = 128; st > 0; st >>= 1) {
        if (threadIdx.x < st) { s1[threadIdx.x] += s1[threadIdx.x + st]; s2[threadIdx.x] += s2[threadIdx.x + st]; }
        __syncthreads();
    }
    if (threadIdx.x == 0) {
        float mu = s1[0] / (float)kL;
        float var = s2[0] / (float)kL - mu * mu;
        mu_rstd[bc] = mu;
        mu_rstd[kB * kCIN + bc] = rsqrtf(var + kEPS);
    }
}

// ---------------- RevIN + patch embedding + pos embed -> x0 (B,P,E) f32 ----------------
__global__ void patch_embed_kernel(const float* __restrict__ imu,
                                   const float* __restrict__ mu_rstd,
                                   const float* __restrict__ revin_w,
                                   const float* __restrict__ revin_b,
                                   const float* __restrict__ patch_w,
                                   const float* __restrict__ patch_b,
                                   const float* __restrict__ pos_embed,
                                   float* __restrict__ x0) {
    int bp = blockIdx.x;                 // 0..BP-1
    int b = bp / kP, p = bp % kP;
    int e = threadIdx.x;                 // 0..127
    float acc = patch_b[e];
    #pragma unroll
    for (int c = 0; c < kCIN; ++c) {
        float mu = mu_rstd[b * kCIN + c];
        float rs = mu_rstd[kB * kCIN + b * kCIN + c];
        float scale = rs * revin_w[c];
        float shift = revin_b[c] - mu * scale;
        #pragma unroll
        for (int k = 0; k < kPATCH; ++k) {
            float v = imu[((size_t)b * kL + p * kPATCH + k) * kCIN + c] * scale + shift;
            acc += v * patch_w[(e * kCIN + c) * kPATCH + k];
        }
    }
    acc += pos_embed[p * kE + e];
    x0[(size_t)bp * kE + e] = acc;
}

// ---------------- res update: res = first ? h : res + h ----------------
__global__ void res_update_kernel(float* __restrict__ res, const float* __restrict__ h,
                                  int n, int first) {
    int i = blockIdx.x * 256 + threadIdx.x;
    if (i < n) res[i] = first ? h[i] : (res[i] + h[i]);
}

// ---------------- LayerNorm pair -> f16; z=1 reads flipped p ----------------
__global__ void ln_pair_kernel(const float* __restrict__ res,
                               const float* __restrict__ ln_w,
                               const float* __restrict__ ln_b,
                               _Float16* __restrict__ hn16, int layer0) {
    int z = blockIdx.y;
    int bp = blockIdx.x;
    int b = bp / kP, p = bp % kP;
    int psrc = z ? (kP - 1 - p) : p;
    int l = layer0 + z;
    const float* x = res + ((size_t)b * kP + psrc) * kE;
    int t = threadIdx.x;
    float v = x[t];
    __shared__ float s1[kE], s2[kE];
    s1[t] = v; s2[t] = v * v;
    __syncthreads();
    for (int s = 64; s > 0; s >>= 1) {
        if (t < s) { s1[t] += s1[t + s]; s2[t] += s2[t + s]; }
        __syncthreads();
    }
    float mu = s1[0] * (1.0f / kE);
    float var = s2[0] * (1.0f / kE) - mu * mu;
    float r = rsqrtf(var + kEPS);
    float o = (v - mu) * r * ln_w[l * kE + t] + ln_b[l * kE + t];
    hn16[((size_t)z * kBP + bp) * kE + t] = (_Float16)o;
}

// ---------------- fragment loaders (ISA 16-bit A 16x32 / B 32x16 layouts) ----------------
__device__ __forceinline__ v16h load_a_frag(const _Float16* __restrict__ row, int k0, int kbaseA) {
    const h8v* ap = (const h8v*)(row + k0 + kbaseA);
    h8v lo = ap[0];
    h8v hi = ap[2];                      // +16 halfs
    v16h a;
    #pragma unroll
    for (int j = 0; j < 8; ++j) { a[j] = lo[j]; a[j + 8] = hi[j]; }
    return a;
}
__device__ __forceinline__ v16h load_b_frag(const _Float16* __restrict__ row, int k0, int kbaseB) {
    const h8v* bp = (const h8v*)(row + k0 + kbaseB);
    h8v b0 = bp[0];
    h8v b1 = bp[2];
    v16h b;
    #pragma unroll
    for (int j = 0; j < 8; ++j) { b[j] = b0[j]; b[j + 8] = b1[j]; }
    return b;
}

// ---- WMMA GEMM, 2x2 register blocking: C[M,N] = A[M,K] * B[N,K]^T (f16 in, f32 out) ----
// Each wave computes a 32x32 block (4 accumulators); per K-step: 2 A-frags + 2 B-frags
// feed 4 v_wmma -> 1 WMMA per fragment load. grid: (M/32, (N/32)/4, 2); block: 128.
__global__ void wmma_gemm_nt_kernel(const _Float16* __restrict__ A,
                                    const _Float16* __restrict__ Bw,
                                    float* __restrict__ C,
                                    int N, int K,
                                    size_t strideAz, size_t strideBz, size_t strideCz) {
    int z = blockIdx.z;
    A  += (size_t)z * strideAz;
    Bw += (size_t)z * strideBz;
    C  += (size_t)z * strideCz;
    int wave = threadIdx.x >> 5;
    int lane = threadIdx.x & 31;
    int mt2 = blockIdx.x;                // 32-row block index
    int nt2 = blockIdx.y * 4 + wave;     // 32-col block index
    int mrow0 = mt2 * 32 + (lane & 15);
    int mrow1 = mrow0 + 16;
    int ncol0 = nt2 * 32 + (lane & 15);
    int ncol1 = ncol0 + 16;
    int kbaseA = (lane < 16) ? 0 : 8;
    int kbaseB = (lane < 16) ? 0 : 8;
    const _Float16* arow0 = A + (size_t)mrow0 * K;
    const _Float16* arow1 = A + (size_t)mrow1 * K;
    const _Float16* brow0 = Bw + (size_t)ncol0 * K;
    const _Float16* brow1 = Bw + (size_t)ncol1 * K;
    v8f acc00 = {}, acc01 = {}, acc10 = {}, acc11 = {};
    for (int k0 = 0; k0 < K; k0 += 32) {
        v16h a0 = load_a_frag(arow0, k0, kbaseA);
        v16h a1 = load_a_frag(arow1, k0, kbaseA);
        v16h b0 = load_b_frag(brow0, k0, kbaseB);
        v16h b1 = load_b_frag(brow1, k0, kbaseB);
        acc00 = __builtin_amdgcn_wmma_f32_16x16x32_f16(false, a0, false, b0, (short)0, acc00, false, false);
        acc01 = __builtin_amdgcn_wmma_f32_16x16x32_f16(false, a0, false, b1, (short)0, acc01, false, false);
        acc10 = __builtin_amdgcn_wmma_f32_16x16x32_f16(false, a1, false, b0, (short)0, acc10, false, false);
        acc11 = __builtin_amdgcn_wmma_f32_16x16x32_f16(false, a1, false, b1, (short)0, acc11, false, false);
    }
    int rb0 = mt2 * 32 + ((lane < 16) ? 0 : 8);  // C tile layout: VGPR r -> row r (+8 for hi lanes)
    int rb1 = rb0 + 16;
    int c0 = nt2 * 32 + (lane & 15);
    int c1 = c0 + 16;
    #pragma unroll
    for (int r = 0; r < 8; ++r) {
        C[(size_t)(rb0 + r) * N + c0] = acc00[r];
        C[(size_t)(rb0 + r) * N + c1] = acc01[r];
        C[(size_t)(rb1 + r) * N + c0] = acc10[r];
        C[(size_t)(rb1 + r) * N + c1] = acc11[r];
    }
}

// ---------------- depthwise causal conv (k=4) + bias + SiLU ----------------
__global__ void conv_silu_kernel(const float* __restrict__ xz,
                                 const float* __restrict__ conv_w,
                                 const float* __restrict__ conv_b,
                                 float* __restrict__ xcf, int layer0) {
    int z = blockIdx.y;
    int bp = blockIdx.x;
    int b = bp / kP, p = bp % kP;
    int d = threadIdx.x;                 // 0..255
    int l = layer0 + z;
    float acc = conv_b[l * kDI + d];
    #pragma unroll
    for (int k = 0; k < kDCONV; ++k) {
        int pp = p - (kDCONV - 1) + k;
        if (pp >= 0)
            acc += xz[((size_t)z * kBP + b * kP + pp) * (2 * kDI) + d] *
                   conv_w[((size_t)l * kDI + d) * kDCONV + k];
    }
    xcf[((size_t)z * kBP + bp) * kDI + d] = siluf(acc);
}

// ---------------- x-projection: dbl[bp, 40] = xc (256) . xproj_w[40,256] ----------------
__global__ void xproj_kernel(const float* __restrict__ xcf,
                             const float* __restrict__ xproj_w,
                             float* __restrict__ dbl, int layer0) {
    int z = blockIdx.y;
    int bp = blockIdx.x;
    int l = layer0 + z;
    int t = threadIdx.x;                 // 0..63
    __shared__ float xs[kDI];
    for (int i = t; i < kDI; i += 64) xs[i] = xcf[((size_t)z * kBP + bp) * kDI + i];
    __syncthreads();
    if (t < kDR + 2 * kDS) {
        const float* wr = xproj_w + ((size_t)l * (kDR + 2 * kDS) + t) * kDI;
        float a = 0.f;
        #pragma unroll 8
        for (int i = 0; i < kDI; ++i) a += xs[i] * wr[i];
        dbl[((size_t)z * kBP + bp) * (kDR + 2 * kDS) + t] = a;
    }
}

// ---------------- dt = softplus(dt_lowrank @ dt_w^T + dt_b) ----------------
__global__ void dtproj_kernel(const float* __restrict__ dbl,
                              const float* __restrict__ dt_w,
                              const float* __restrict__ dt_b,
                              float* __restrict__ dtbuf, int layer0) {
    int z = blockIdx.y;
    int bp = blockIdx.x;
    int l = layer0 + z;
    int d = threadIdx.x;                 // 0..255
    __shared__ float ds[kDR];
    if (d < kDR) ds[d] = dbl[((size_t)z * kBP + bp) * (kDR + 2 * kDS) + d];
    __syncthreads();
    float a = dt_b[l * kDI + d];
    #pragma unroll
    for (int r = 0; r < kDR; ++r)
        a += ds[r] * dt_w[((size_t)l * kDI + d) * kDR + r];
    dtbuf[((size_t)z * kBP + bp) * kDI + d] = softplusf(a);
}

// ---------------- selective scan over P + skip + gate -> y16 (f16) ----------------
__global__ void scan_kernel(const float* __restrict__ dtbuf,
                            const float* __restrict__ dbl,
                            const float* __restrict__ xcf,
                            const float* __restrict__ xz,
                            const float* __restrict__ A_log,
                            const float* __restrict__ Dp,
                            _Float16* __restrict__ y16, int layer0) {
    int z = blockIdx.y;
    int b = blockIdx.x;
    int d = threadIdx.x;                 // 0..255
    int l = layer0 + z;
    float Ac[kDS];
    #pragma unroll
    for (int n = 0; n < kDS; ++n)
        Ac[n] = -__expf(A_log[((size_t)l * kDI + d) * kDS + n]);
    float Dv = Dp[l * kDI + d];
    float hreg[kDS];
    #pragma unroll
    for (int n = 0; n < kDS; ++n) hreg[n] = 0.f;
    __shared__ float Bs[kDS], Cs[kDS];
    for (int p = 0; p < kP; ++p) {
        size_t rb = (size_t)z * kBP + (size_t)b * kP + p;
        __syncthreads();
        if (d < kDS)            Bs[d]        = dbl[rb * (kDR + 2 * kDS) + kDR + d];
        else if (d < 2 * kDS)   Cs[d - kDS]  = dbl[rb * (kDR + 2 * kDS) + kDR + kDS + (d - kDS)];
        __syncthreads();
        float dt = dtbuf[rb * kDI + d];
        float xv = xcf[rb * kDI + d];
        float y = 0.f;
        #pragma unroll
        for (int n = 0; n < kDS; ++n) {
            hreg[n] = __expf(dt * Ac[n]) * hreg[n] + dt * Bs[n] * xv;
            y += hreg[n] * Cs[n];
        }
        float zv = xz[rb * (2 * kDI) + kDI + d];
        float out = (y + Dv * xv) * siluf(zv);
        y16[rb * kDI + d] = (_Float16)out;
    }
}

// ---------------- merge: h = out_f + flip(out_b); res *= 2 ----------------
__global__ void merge_kernel(const float* __restrict__ outbuf,
                             float* __restrict__ h, float* __restrict__ res) {
    int bp = blockIdx.x;
    int b = bp / kP, p = bp % kP;
    int e = threadIdx.x;
    size_t i0 = (size_t)bp * kE + e;
    size_t i1 = ((size_t)kBP + (size_t)b * kP + (kP - 1 - p)) * kE + e;
    h[(size_t)bp * kE + e] = outbuf[i0] + outbuf[i1];
    res[(size_t)bp * kE + e] *= 2.0f;
}

// ---------------- final layernorm of (h + res) ----------------
__global__ void final_ln_kernel(const float* __restrict__ h, const float* __restrict__ res,
                                const float* __restrict__ w, const float* __restrict__ bb,
                                float* __restrict__ out) {
    int bp = blockIdx.x;
    int t = threadIdx.x;
    float v = h[(size_t)bp * kE + t] + res[(size_t)bp * kE + t];
    __shared__ float s1[kE], s2[kE];
    s1[t] = v; s2[t] = v * v;
    __syncthreads();
    for (int s = 64; s > 0; s >>= 1) {
        if (t < s) { s1[t] += s1[t + s]; s2[t] += s2[t + s]; }
        __syncthreads();
    }
    float mu = s1[0] * (1.0f / kE);
    float var = s2[0] * (1.0f / kE) - mu * mu;
    float r = rsqrtf(var + kEPS);
    out[(size_t)bp * kE + t] = (v - mu) * r * w[t] + bb[t];
}

// ---------------- mean over P ----------------
__global__ void colmean_kernel(const float* __restrict__ t, float* __restrict__ feat) {
    int b = blockIdx.x;
    int e = threadIdx.x;
    float a = 0.f;
    for (int p = 0; p < kP; ++p) a += t[((size_t)b * kP + p) * kE + e];
    feat[(size_t)b * kE + e] = a * (1.0f / kP);
}

// ---------------- classification head ----------------
__global__ void head_kernel(const float* __restrict__ feat,
                            const float* __restrict__ head_w,
                            const float* __restrict__ head_b,
                            float* __restrict__ out) {
    int t = threadIdx.x;                 // 0..287
    if (t >= kB * kNCLS) return;
    int b = t / kNCLS, c = t % kNCLS;
    float a = head_b[c];
    #pragma unroll 8
    for (int e = 0; e < kE; ++e) a += feat[(size_t)b * kE + e] * head_w[c * kE + e];
    out[(size_t)b * kNCLS + c] = a;
}

// ================================================================
extern "C" void kernel_launch(void* const* d_in, const int* in_sizes, int n_in,
                              void* d_out, int out_size, void* d_ws, size_t ws_size,
                              hipStream_t stream) {
    (void)in_sizes; (void)n_in; (void)out_size; (void)ws_size;
    const float* imu      = (const float*)d_in[0];
    const float* revin_w  = (const float*)d_in[1];
    const float* revin_b  = (const float*)d_in[2];
    const float* patch_w  = (const float*)d_in[3];
    const float* patch_b  = (const float*)d_in[4];
    const float* pos_emb  = (const float*)d_in[5];
    const float* ln_w     = (const float*)d_in[6];
    const float* ln_b     = (const float*)d_in[7];
    const float* in_w     = (const float*)d_in[8];
    const float* conv_w   = (const float*)d_in[9];
    const float* conv_b   = (const float*)d_in[10];
    const float* xproj_w  = (const float*)d_in[11];
    const float* dt_w     = (const float*)d_in[12];
    const float* dt_b     = (const float*)d_in[13];
    const float* A_log    = (const float*)d_in[14];
    const float* Dp       = (const float*)d_in[15];
    const float* out_w    = (const float*)d_in[16];
    const float* normf_w  = (const float*)d_in[17];
    const float* normf_b  = (const float*)d_in[18];
    const float* head_w   = (const float*)d_in[19];
    const float* head_b   = (const float*)d_in[20];

    // ---- workspace bump allocator (256B aligned) ----
    char* base = (char*)d_ws;
    auto alloc = [&](size_t bytes) -> void* {
        void* p = (void*)base;
        base += (bytes + 255) & ~(size_t)255;
        return p;
    };
    const int nIW = kDEPTH * 2 * kDI * kE;   // 786432
    const int nOW = kDEPTH * kE * kDI;       // 393216
    _Float16* w16_in  = (_Float16*)alloc((size_t)nIW * 2);
    _Float16* w16_out = (_Float16*)alloc((size_t)nOW * 2);
    float* mu_rstd    = (float*)alloc(2 * kB * kCIN * 4);
    float* hbuf       = (float*)alloc((size_t)kBP * kE * 4);
    float* resbuf     = (float*)alloc((size_t)kBP * kE * 4);
    _Float16* hn16    = (_Float16*)alloc((size_t)2 * kBP * kE * 2);
    float* xz         = (float*)alloc((size_t)2 * kBP * 2 * kDI * 4);
    float* xcf        = (float*)alloc((size_t)2 * kBP * kDI * 4);
    float* dbl        = (float*)alloc((size_t)2 * kBP * (kDR + 2 * kDS) * 4);
    float* dtbuf      = (float*)alloc((size_t)2 * kBP * kDI * 4);
    _Float16* y16     = (_Float16*)alloc((size_t)2 * kBP * kDI * 2);
    float* outbuf     = (float*)alloc((size_t)2 * kBP * kE * 4);
    float* tmpln      = (float*)alloc((size_t)kBP * kE * 4);
    float* feat       = (float*)alloc((size_t)kB * kE * 4);

    // ---- weight conversion to f16 ----
    cvt16_kernel<<<(nIW + 255) / 256, 256, 0, stream>>>(in_w, w16_in, nIW);
    cvt16_kernel<<<(nOW + 255) / 256, 256, 0, stream>>>(out_w, w16_out, nOW);

    // ---- stem ----
    revin_stats_kernel<<<kB * kCIN, 256, 0, stream>>>(imu, mu_rstd);
    patch_embed_kernel<<<kBP, kE, 0, stream>>>(imu, mu_rstd, revin_w, revin_b,
                                               patch_w, patch_b, pos_emb, hbuf);

    const int NHE = kBP * kE;
    for (int i = 0; i < kDEPTH / 2; ++i) {
        int l0 = 2 * i;
        res_update_kernel<<<(NHE + 255) / 256, 256, 0, stream>>>(resbuf, hbuf, NHE, i == 0);
        ln_pair_kernel<<<dim3(kBP, 2), kE, 0, stream>>>(resbuf, ln_w, ln_b, hn16, l0);
        // in_proj: M=2048, N=512, K=128 -> (M/32, (N/32)/4, 2)
        wmma_gemm_nt_kernel<<<dim3(kBP / 32, (2 * kDI) / 128, 2), 128, 0, stream>>>(
            hn16, w16_in + (size_t)l0 * 2 * kDI * kE, xz,
            2 * kDI, kE,
            (size_t)kBP * kE, (size_t)2 * kDI * kE, (size_t)kBP * 2 * kDI);
        conv_silu_kernel<<<dim3(kBP, 2), kDI, 0, stream>>>(xz, conv_w, conv_b, xcf, l0);
        xproj_kernel<<<dim3(kBP, 2), 64, 0, stream>>>(xcf, xproj_w, dbl, l0);
        dtproj_kernel<<<dim3(kBP, 2), kDI, 0, stream>>>(dbl, dt_w, dt_b, dtbuf, l0);
        scan_kernel<<<dim3(kB, 2), kDI, 0, stream>>>(dtbuf, dbl, xcf, xz, A_log, Dp, y16, l0);
        // out_proj: M=2048, N=128, K=256 -> (M/32, (N/32)/4, 2) = (64, 1, 2)
        wmma_gemm_nt_kernel<<<dim3(kBP / 32, kE / 128, 2), 128, 0, stream>>>(
            y16, w16_out + (size_t)l0 * kE * kDI, outbuf,
            kE, kDI,
            (size_t)kBP * kDI, (size_t)kE * kDI, (size_t)kBP * kE);
        merge_kernel<<<kBP, kE, 0, stream>>>(outbuf, hbuf, resbuf);
    }

    final_ln_kernel<<<kBP, kE, 0, stream>>>(hbuf, resbuf, normf_w, normf_b, tmpln);
    colmean_kernel<<<kB, kE, 0, stream>>>(tmpln, feat);
    head_kernel<<<1, 288, 0, stream>>>(feat, head_w, head_b, (float*)d_out);
}
